// MoE_807453852457
// MI455X (gfx1250) — matrix operand order repaired and checked by reading, not compiled, not run on hardware
//
#include <hip/hip_runtime.h>
#include <math.h>

#define DMODEL 1024
#define NEXP   32
#define FF     512
#define TOPK   4

typedef __attribute__((ext_vector_type(16))) __bf16 bf16x16;
typedef __attribute__((ext_vector_type(8)))  float  f32x8;

struct BF2 { __bf16 hi; __bf16 lo; };

__device__ __forceinline__ BF2 split_bf16(float f) {
    BF2 r;
    r.hi = (__bf16)f;
    r.lo = (__bf16)(f - (float)r.hi);
    return r;
}

#define WMMA_BF16(A, B, C) \
    __builtin_amdgcn_wmma_f32_16x16x32_bf16(false, (A), false, (B), (short)0, (C), false, false)

// -------------------------------------------------------------------------
// Routing: one wave per token. Lane e computes logit for expert e, sigmoid,
// then 4 rounds of cross-lane argmax (ties -> lower index, matches top_k).
// -------------------------------------------------------------------------
__global__ __launch_bounds__(256)
void moe_route(const float* __restrict__ x, const float* __restrict__ sel,
               int* __restrict__ cnt, int* __restrict__ tokList,
               float* __restrict__ gateList, int nTok) {
    const int wave = threadIdx.x >> 5;
    const int lane = threadIdx.x & 31;
    const int t = blockIdx.x * 8 + wave;
    if (t >= nTok) return;

    const float* xr = x + (size_t)t * DMODEL;
    const float* sr = sel + (size_t)lane * DMODEL;
    float acc = 0.f;
    for (int d = 0; d < DMODEL; d += 4) {
        float4 xv = *(const float4*)(xr + d);
        float4 sv = *(const float4*)(sr + d);
        acc += xv.x * sv.x + xv.y * sv.y + xv.z * sv.z + xv.w * sv.w;
    }
    const float gate = 1.f / (1.f + __expf(-acc));
    float v = gate;

    for (int k = 0; k < TOPK; ++k) {
        float bv = v; int bi = lane;
        #pragma unroll
        for (int off = 16; off; off >>= 1) {
            float ov = __shfl_xor(bv, off);
            int   oi = __shfl_xor(bi, off);
            if (ov > bv || (ov == bv && oi < bi)) { bv = ov; bi = oi; }
        }
        if (lane == bi) {
            v = -1.f;
            int pos = atomicAdd(&cnt[bi], 1);
            tokList[(size_t)bi * nTok + pos]  = t;
            gateList[(size_t)bi * nTok + pos] = bv;
        }
    }
}

// -------------------------------------------------------------------------
// Pre-swizzle a weight matrix set W[nExp][Krows][Ncols] (fp32, row-major)
// into WMMA B-fragment layout, split into bf16 hi/lo.
// Fragment (e, kb, nt): lane holds 16 bf16 at K = i + 8*(i>>3) + 8*g, N = nt*16 + l.
// Blob layout: frag-major, lane-major, 16 contiguous bf16 per lane.
// -------------------------------------------------------------------------
__global__ __launch_bounds__(256)
void swizzle_weights(const float* __restrict__ W, __bf16* __restrict__ hi,
                     __bf16* __restrict__ lo, int Krows, int Ncols, int nExp) {
    size_t id  = (size_t)blockIdx.x * 256 + threadIdx.x;
    int lane   = (int)(id & 31);
    size_t fid = id >> 5;
    const int ntn = Ncols >> 4;
    const int nkb = Krows >> 5;
    const size_t nfrag = (size_t)nExp * nkb * ntn;
    if (fid >= nfrag) return;
    int nt = (int)(fid % ntn);
    size_t tmp = fid / ntn;
    int kb = (int)(tmp % nkb);
    int e  = (int)(tmp / nkb);
    const int l = lane & 15, g = lane >> 4;

    const float* src = W + ((size_t)e * Krows + (size_t)kb * 32) * Ncols + nt * 16 + l;
    bf16x16 vhi, vlo;
    #pragma unroll
    for (int i = 0; i < 16; ++i) {
        int K = i + ((i >> 3) << 3) + g * 8;
        float f = src[(size_t)K * Ncols];
        BF2 s = split_bf16(f);
        vhi[i] = s.hi; vlo[i] = s.lo;
    }
    size_t off = (fid * 32 + lane) * 16;
    *(bf16x16*)(hi + off) = vhi;
    *(bf16x16*)(lo + off) = vlo;
}

// -------------------------------------------------------------------------
// Main expert GEMM, pre-swizzled-weight path.
// Block = (expert e, tile of 16 tokens). 8 waves.
// GEMM1: H = relu(X @ keys[e]) * gate ; GEMM2: out += H @ values[e]
// A fragments staged/converted cooperatively in LDS (bf16 hi/lo, swizzled),
// B fragments are straight b128 loads from the swizzled workspace.
// -------------------------------------------------------------------------
__global__ __launch_bounds__(256)
void moe_expert_sw(const float* __restrict__ x,
                   const __bf16* __restrict__ khi, const __bf16* __restrict__ klo,
                   const __bf16* __restrict__ whi, const __bf16* __restrict__ wlo,
                   const int* __restrict__ tokList, const float* __restrict__ gateList,
                   const int* __restrict__ cnt, float* __restrict__ out, int nTok) {
    const int e    = blockIdx.y;
    const int tile = blockIdx.x;
    const int ne   = cnt[e];
    if (tile * 16 >= ne) return;               // block-uniform

    __shared__ __attribute__((aligned(32))) __bf16 Ahi[4][32][16];   // one 128-wide K chunk
    __shared__ __attribute__((aligned(32))) __bf16 Alo[4][32][16];
    __shared__ __attribute__((aligned(32))) __bf16 Hhi[16][32][16];  // full F=512
    __shared__ __attribute__((aligned(32))) __bf16 Hlo[16][32][16];
    __shared__ int   idx_s[16];
    __shared__ float gate_s[16];

    const int tid  = threadIdx.x;
    const int wv   = tid >> 5;
    const int lane = tid & 31;
    const int l    = lane & 15;
    const int g    = lane >> 4;
    const int base = tile * 16;

    if (tid < 16) {
        int r = base + tid;
        if (r < ne) {
            idx_s[tid]  = tokList[(size_t)e * nTok + r];
            gate_s[tid] = gateList[(size_t)e * nTok + r];
        } else {
            idx_s[tid]  = tokList[(size_t)e * nTok + base];
            gate_s[tid] = 0.f;
        }
    }
    __syncthreads();

    const f32x8 vzero = {0.f,0.f,0.f,0.f,0.f,0.f,0.f,0.f};

    // ---------------- GEMM1: X @ keys[e] -> H ----------------
    const int fbase = wv * 64;                 // wave's 64 F columns (4 tiles)
    f32x8 acc[4] = {vzero, vzero, vzero, vzero};

    const int stage_m  = tid >> 4;             // 0..15
    const int stage_cb = (tid & 15) * 8;       // 0..120

    for (int kc = 0; kc < DMODEL; kc += 128) {
        __syncthreads();
        // cooperative load + convert + swizzle of X chunk into A fragments
        {
            const float* xr = x + (size_t)idx_s[stage_m] * DMODEL + kc + stage_cb;
            #pragma unroll
            for (int j = 0; j < 8; ++j) {
                int c = stage_cb + j;
                BF2 s = split_bf16(xr[j]);
                int kbl = c >> 5, K = c & 31;
                int gg = (K >> 3) & 1;
                int ii = (K & 7) + 8 * (K >> 4);
                Ahi[kbl][stage_m + 16 * gg][ii] = s.hi;
                Alo[kbl][stage_m + 16 * gg][ii] = s.lo;
            }
        }
        __syncthreads();

        for (int kbl = 0; kbl < 4; ++kbl) {
            bf16x16 ah = *(const bf16x16*)&Ahi[kbl][lane][0];
            bf16x16 al = *(const bf16x16*)&Alo[kbl][lane][0];
            const int kb = (kc >> 5) + kbl;    // global K block, 0..31
            #pragma unroll
            for (int ft = 0; ft < 4; ++ft) {
                const int ftg = (fbase >> 4) + ft;            // global N tile, 0..31
                size_t off = ((((size_t)e * 32 + kb) * 32 + ftg) * 32 + lane) * 16;
                bf16x16 bh = *(const bf16x16*)(khi + off);
                bf16x16 bl = *(const bf16x16*)(klo + off);
                acc[ft] = WMMA_BF16(ah, bh, acc[ft]);
                acc[ft] = WMMA_BF16(ah, bl, acc[ft]);
                acc[ft] = WMMA_BF16(al, bh, acc[ft]);
            }
        }
    }

    // relu + gate, store H as swizzled bf16 hi/lo A-fragments
    #pragma unroll
    for (int ft = 0; ft < 4; ++ft) {
        #pragma unroll
        for (int r = 0; r < 8; ++r) {
            int m = r + g * 8;
            int f = fbase + ft * 16 + l;
            float h = acc[ft][r];
            h = (h > 0.f) ? h * gate_s[m] : 0.f;
            BF2 s = split_bf16(h);
            int kbl = f >> 5, K = f & 31;
            int gg = (K >> 3) & 1;
            int ii = (K & 7) + 8 * (K >> 4);
            Hhi[kbl][m + 16 * gg][ii] = s.hi;
            Hlo[kbl][m + 16 * gg][ii] = s.lo;
        }
    }
    __syncthreads();

    // ---------------- GEMM2: H @ values[e] -> out ----------------
    const int nb = wv * 128;                   // wave's 128 D columns (8 tiles)
    f32x8 o[8] = {vzero, vzero, vzero, vzero, vzero, vzero, vzero, vzero};

    for (int kb = 0; kb < 16; ++kb) {
        bf16x16 ah = *(const bf16x16*)&Hhi[kb][lane][0];
        bf16x16 al = *(const bf16x16*)&Hlo[kb][lane][0];
        #pragma unroll
        for (int nt = 0; nt < 8; ++nt) {
            const int ntg = (nb >> 4) + nt;    // global N tile, 0..63
            size_t off = ((((size_t)e * 16 + kb) * 64 + ntg) * 32 + lane) * 16;
            bf16x16 bh = *(const bf16x16*)(whi + off);
            bf16x16 bl = *(const bf16x16*)(wlo + off);
            o[nt] = WMMA_BF16(ah, bh, o[nt]);
            o[nt] = WMMA_BF16(ah, bl, o[nt]);
            o[nt] = WMMA_BF16(al, bh, o[nt]);
        }
    }

    #pragma unroll
    for (int nt = 0; nt < 8; ++nt) {
        #pragma unroll
        for (int r = 0; r < 8; ++r) {
            int m = r + g * 8;
            int t = idx_s[m];
            __hip_atomic_fetch_add(&out[(size_t)t * DMODEL + nb + nt * 16 + l],
                                   o[nt][r], __ATOMIC_RELAXED, __HIP_MEMORY_SCOPE_AGENT);
        }
    }
}

// -------------------------------------------------------------------------
// Fallback path (workspace too small for swizzled weights): convert weights
// to bf16 hi/lo fragments on the fly from fp32 global memory.
// -------------------------------------------------------------------------
__global__ __launch_bounds__(256)
void moe_expert_direct(const float* __restrict__ x, const float* __restrict__ keys,
                       const float* __restrict__ values, const int* __restrict__ tokList,
                       const float* __restrict__ gateList, const int* __restrict__ cnt,
                       float* __restrict__ out, int nTok) {
    const int e    = blockIdx.y;
    const int tile = blockIdx.x;
    const int ne   = cnt[e];
    if (tile * 16 >= ne) return;

    __shared__ float Xs[16][132];
    __shared__ float Hs[16][520];
    __shared__ int   idx_s[16];
    __shared__ float gate_s[16];

    const int tid  = threadIdx.x;
    const int wv   = tid >> 5;
    const int lane = tid & 31;
    const int l    = lane & 15;
    const int g    = lane >> 4;
    const int base = tile * 16;

    if (tid < 16) {
        int r = base + tid;
        if (r < ne) {
            idx_s[tid]  = tokList[(size_t)e * nTok + r];
            gate_s[tid] = gateList[(size_t)e * nTok + r];
        } else {
            idx_s[tid]  = tokList[(size_t)e * nTok + base];
            gate_s[tid] = 0.f;
        }
    }
    __syncthreads();

    const f32x8 vzero = {0.f,0.f,0.f,0.f,0.f,0.f,0.f,0.f};
    const float* Kp = keys + (size_t)e * DMODEL * FF;
    const int fbase = wv * 64;
    f32x8 acc[4] = {vzero, vzero, vzero, vzero};

    for (int kc = 0; kc < DMODEL; kc += 128) {
        __syncthreads();
        for (int i = tid; i < 16 * 128; i += 256) {
            int r = i >> 7, c = i & 127;
            Xs[r][c] = x[(size_t)idx_s[r] * DMODEL + kc + c];
        }
        __syncthreads();

        for (int ks = 0; ks < 128; ks += 32) {
            bf16x16 ahi, alo;
            #pragma unroll
            for (int i = 0; i < 16; ++i) {
                int kk = i + g * 16;
                BF2 s = split_bf16(Xs[l][ks + kk]);
                ahi[i] = s.hi; alo[i] = s.lo;
            }
            #pragma unroll
            for (int ft = 0; ft < 4; ++ft) {
                bf16x16 bhi, blo;
                #pragma unroll
                for (int i = 0; i < 16; ++i) {
                    int kk = i + g * 16;
                    BF2 s = split_bf16(Kp[(size_t)(kc + ks + kk) * FF + fbase + ft * 16 + l]);
                    bhi[i] = s.hi; blo[i] = s.lo;
                }
                acc[ft] = WMMA_BF16(ahi, bhi, acc[ft]);
                acc[ft] = WMMA_BF16(ahi, blo, acc[ft]);
                acc[ft] = WMMA_BF16(alo, bhi, acc[ft]);
            }
        }
    }

    #pragma unroll
    for (int ft = 0; ft < 4; ++ft) {
        #pragma unroll
        for (int r = 0; r < 8; ++r) {
            int m = r + g * 8;
            float h = acc[ft][r];
            h = (h > 0.f) ? h * gate_s[m] : 0.f;
            Hs[m][fbase + ft * 16 + l] = h;
        }
    }
    __syncthreads();

    const float* Vp = values + (size_t)e * FF * DMODEL;
    const int nb = wv * 128;
    f32x8 o[8] = {vzero, vzero, vzero, vzero, vzero, vzero, vzero, vzero};

    for (int ks = 0; ks < FF; ks += 32) {
        bf16x16 ahi, alo;
        #pragma unroll
        for (int i = 0; i < 16; ++i) {
            int kk = i + g * 16;
            BF2 s = split_bf16(Hs[l][ks + kk]);
            ahi[i] = s.hi; alo[i] = s.lo;
        }
        #pragma unroll
        for (int nt = 0; nt < 8; ++nt) {
            bf16x16 bhi, blo;
            #pragma unroll
            for (int i = 0; i < 16; ++i) {
                int kk = i + g * 16;
                BF2 s = split_bf16(Vp[(size_t)(ks + kk) * DMODEL + nb + nt * 16 + l]);
                bhi[i] = s.hi; blo[i] = s.lo;
            }
            o[nt] = WMMA_BF16(ahi, bhi, o[nt]);
            o[nt] = WMMA_BF16(ahi, blo, o[nt]);
            o[nt] = WMMA_BF16(alo, bhi, o[nt]);
        }
    }

    #pragma unroll
    for (int nt = 0; nt < 8; ++nt) {
        #pragma unroll
        for (int r = 0; r < 8; ++r) {
            int m = r + g * 8;
            int t = idx_s[m];
            __hip_atomic_fetch_add(&out[(size_t)t * DMODEL + nb + nt * 16 + l],
                                   o[nt][r], __ATOMIC_RELAXED, __HIP_MEMORY_SCOPE_AGENT);
        }
    }
}

extern "C" void kernel_launch(void* const* d_in, const int* in_sizes, int n_in,
                              void* d_out, int out_size, void* d_ws, size_t ws_size,
                              hipStream_t stream) {
    const float* x      = (const float*)d_in[0];
    const float* sel    = (const float*)d_in[1];
    const float* keys   = (const float*)d_in[2];
    const float* values = (const float*)d_in[3];
    float* out = (float*)d_out;

    const int nTok = in_sizes[0] / DMODEL;   // 4096

    // ---- workspace layout (byte offsets, 256-aligned sections) ----
    char* ws = (char*)d_ws;
    size_t off = 0;
    auto take = [&](size_t bytes) { size_t o = off; off = (off + bytes + 255) & ~(size_t)255; return o; };

    int*   cnt      = (int*)(ws + take(NEXP * sizeof(int)));
    int*   tokList  = (int*)(ws + take((size_t)NEXP * nTok * sizeof(int)));
    float* gateList = (float*)(ws + take((size_t)NEXP * nTok * sizeof(float)));

    const size_t keyElems = (size_t)NEXP * DMODEL * FF;
    const size_t valElems = (size_t)NEXP * FF * DMODEL;
    size_t o_khi = take(keyElems * sizeof(__bf16));
    size_t o_klo = take(keyElems * sizeof(__bf16));
    size_t o_whi = take(valElems * sizeof(__bf16));
    size_t o_wlo = take(valElems * sizeof(__bf16));
    const bool useSwizzle = (off <= ws_size);

    hipMemsetAsync(cnt, 0, NEXP * sizeof(int), stream);
    hipMemsetAsync(out, 0, (size_t)out_size * sizeof(float), stream);

    const int routeBlocks = (nTok + 7) / 8;
    moe_route<<<routeBlocks, 256, 0, stream>>>(x, sel, cnt, tokList, gateList, nTok);

    const int maxTiles = (nTok + 15) / 16;
    dim3 grid(maxTiles, NEXP);

    if (useSwizzle) {
        __bf16* khi = (__bf16*)(ws + o_khi);
        __bf16* klo = (__bf16*)(ws + o_klo);
        __bf16* whi = (__bf16*)(ws + o_whi);
        __bf16* wlo = (__bf16*)(ws + o_wlo);
        // keys: 32 x (1024 K -> 32 kb) x (512 N -> 32 tiles)
        {
            size_t threads = (size_t)NEXP * (DMODEL / 32) * (FF / 16) * 32;
            int blocks = (int)((threads + 255) / 256);
            swizzle_weights<<<blocks, 256, 0, stream>>>(keys, khi, klo, DMODEL, FF, NEXP);
        }
        // values: 32 x (512 K -> 16 kb) x (1024 N -> 64 tiles)
        {
            size_t threads = (size_t)NEXP * (FF / 32) * (DMODEL / 16) * 32;
            int blocks = (int)((threads + 255) / 256);
            swizzle_weights<<<blocks, 256, 0, stream>>>(values, whi, wlo, FF, DMODEL, NEXP);
        }
        moe_expert_sw<<<grid, 256, 0, stream>>>(x, khi, klo, whi, wlo,
                                                tokList, gateList, cnt, out, nTok);
    } else {
        moe_expert_direct<<<grid, 256, 0, stream>>>(x, keys, values,
                                                    tokList, gateList, cnt, out, nTok);
    }
}
